// DeepInterestNetwork_40853728919898
// MI455X (gfx1250) — compile-verified, matrix operand order checked by reading, not run on hardware
//
#include <hip/hip_runtime.h>
#include <hip/hip_bf16.h>
#include <math.h>
#include <stdint.h>

// ---------------------------------------------------------------------------
// DIN forward for MI455X (gfx1250): bf16 WMMA (v_wmma_f32_16x16x32_bf16),
// per-batch folded LAU weights (4x FLOP cut), async global->LDS staging,
// fragment-major packed weights, fast-rcp sigmoid epilogues.
// ---------------------------------------------------------------------------

typedef __attribute__((ext_vector_type(16))) __bf16 v16bf;
typedef __attribute__((ext_vector_type(8)))  __bf16 v8bf;
typedef __attribute__((ext_vector_type(8)))  float  v8f;

#define BDIM 1024
#define TDIM 200
#define EDIM 64
#define HC   128      // 2E
#define DMLP 448      // 7E
#define NEGV (-2147483648.0f)

// sigmoid via v_exp_f32 + v_rcp_f32 (avoids the IEEE divide expansion)
static __device__ inline float sigmoidf_(float x) {
  return __builtin_amdgcn_rcpf(1.0f + __expf(-x));
}

// A-fragment (16x32 bf16, MxK) load from a row-major tile in LDS.
// Lane l (l<16): row M=l, K = k0+{0..7} and k0+{16..23}
// Lane l (>=16): row M=l-16, K = k0+{8..15} and k0+{24..31}
static __device__ inline v16bf load_a_frag(const __bf16* tile, int ld, int k0, int lane) {
  int m  = lane & 15;
  int hi = (lane >> 4) * 8;
  const __bf16* p = tile + m * ld + k0 + hi;
  v8bf lo  = *(const v8bf*)(p);
  v8bf hh  = *(const v8bf*)(p + 16);
  v16bf a;
#pragma unroll
  for (int i = 0; i < 8; ++i) { a[i] = lo[i]; a[i + 8] = hh[i]; }
  return a;
}

static __device__ inline v8f wmma_bf16(v16bf a, v16bf b, v8f c) {
  return __builtin_amdgcn_wmma_f32_16x16x32_bf16(false, a, false, b, (short)0, c,
                                                 false, false);
}

// ---------------------------------------------------------------------------
// K0: gather hist rows [B*T, 128] = [mat_table[hmid] | cat_table[hcat]] -> bf16
// ---------------------------------------------------------------------------
__global__ void k_gather_hist(const int* __restrict__ hmid,
                              const int* __restrict__ hcat,
                              const float* __restrict__ matT,
                              const float* __restrict__ catT,
                              __bf16* __restrict__ histBF) {
  int idx = blockIdx.x * blockDim.x + threadIdx.x;   // [0, B*T*32)
  if (idx >= BDIM * TDIM * 32) return;
  int row = idx >> 5;
  int q   = idx & 31;
  int c4  = (q & 15) * 4;
  const float* src = (q < 16) ? (matT + (size_t)hmid[row] * EDIM + c4)
                              : (catT + (size_t)hcat[row] * EDIM + c4);
  __bf16* dst = histBF + (size_t)row * HC + ((q < 16) ? c4 : (64 + c4));
#pragma unroll
  for (int i = 0; i < 4; ++i) dst[i] = (__bf16)src[i];
}

// K0b: gather current item row [B,128] in f32 and bf16
__global__ void k_gather_cur(const int* __restrict__ mid,
                             const int* __restrict__ cat,
                             const float* __restrict__ matT,
                             const float* __restrict__ catT,
                             float* __restrict__ curF,
                             __bf16* __restrict__ curB) {
  int i = blockIdx.x * blockDim.x + threadIdx.x;
  if (i >= BDIM * HC) return;
  int b = i >> 7, j = i & 127;
  float v = (j < 64) ? matT[(size_t)mid[b] * EDIM + j]
                     : catT[(size_t)cat[b] * EDIM + (j - 64)];
  curF[i] = v;
  curB[i] = (__bf16)v;
}

// ---------------------------------------------------------------------------
// K1: pack fp32 W[K,N] -> WMMA B-fragment-major bf16:
// pk[((kc*nT + nt)*32 + lane)*16 + j]  (one contiguous 32B load per lane)
// ---------------------------------------------------------------------------
__global__ void k_pack_w(const float* __restrict__ W, __bf16* __restrict__ pk,
                         int K, int N) {
  int p = blockIdx.x * blockDim.x + threadIdx.x;
  if (p >= K * N) return;
  int j    = p & 15;
  int lane = (p >> 4) & 31;
  int rest = p >> 9;
  int nT   = N >> 4;
  int nt   = rest % nT;
  int kc   = rest / nT;
  int hi   = (lane >> 4) * 8;
  int n    = lane & 15;
  int k    = kc * 32 + hi + j + ((j >> 3) << 3);
  pk[p] = (__bf16)W[(size_t)k * N + nt * 16 + n];
}

// K1b: pack (W1a + W1c) = lau_w1 rows [0,128) + rows [256,384) -> bf16 frags
__global__ void k_pack_w1ac(const float* __restrict__ W1,
                            __bf16* __restrict__ pk) {
  int p = blockIdx.x * blockDim.x + threadIdx.x;
  if (p >= 128 * 256) return;           // K=128, N=256, nT=16
  int j    = p & 15;
  int lane = (p >> 4) & 31;
  int rest = p >> 9;
  int nt   = rest & 15;
  int kc   = rest >> 4;
  int hi   = (lane >> 4) * 8;
  int n    = lane & 15;
  int k    = kc * 32 + hi + j + ((j >> 3) << 3);
  int col  = nt * 16 + n;
  pk[p] = (__bf16)(W1[(size_t)k * 256 + col] + W1[(size_t)(k + 256) * 256 + col]);
}

// K1c: per-batch effective LAU weight, packed B-fragment bf16:
// W_eff[b][k][n] = (W1b - W1c)[k][n] + cur[b][k] * W1d[k][n]
// layout: pk[b*32768 + ((kc*16 + nt)*32 + lane)*16 + j]
__global__ void k_build_weff(const float* __restrict__ W1,
                             const float* __restrict__ curF,
                             __bf16* __restrict__ pk) {
  int p = blockIdx.x * blockDim.x + threadIdx.x;   // < B * 32768
  int b = p >> 15, q = p & 32767;
  int j    = q & 15;
  int lane = (q >> 4) & 31;
  int rest = q >> 9;
  int nt   = rest & 15;
  int kc   = rest >> 4;
  int hi   = (lane >> 4) * 8;
  int n    = lane & 15;
  int k    = kc * 32 + hi + j + ((j >> 3) << 3);
  int col  = nt * 16 + n;
  float w = (W1[(size_t)(128 + k) * 256 + col] - W1[(size_t)(256 + k) * 256 + col])
          + curF[b * HC + k] * W1[(size_t)(384 + k) * 256 + col];
  pk[p] = (__bf16)w;
}

// ---------------------------------------------------------------------------
// K2: fused LAU + softmax + pooling. One workgroup (256 thr / 8 waves) per b.
// hist tile staged via async global->LDS; GEMM1 uses folded per-b weights.
// ---------------------------------------------------------------------------
__global__ void __launch_bounds__(256) k_lau(
    const int* __restrict__ uid, const int* __restrict__ maskG,
    const float* __restrict__ userT, const __bf16* __restrict__ histBF,
    const float* __restrict__ curFG, const __bf16* __restrict__ weffPk,
    const float* __restrict__ c1, const __bf16* __restrict__ w2pk,
    const float* __restrict__ b2, const float* __restrict__ w3,
    const float* __restrict__ b3, float* __restrict__ x) {
  __shared__ __bf16 histT[16 * HC];     // 4 KB (A matrix for GEMM1)
  __shared__ __bf16 h1T[16 * 256];      // 8 KB
  __shared__ __bf16 h2T[16 * HC];       // 4 KB
  __shared__ float  awL[224];
  __shared__ float  sMsum;

  int b    = blockIdx.x;
  int tid  = threadIdx.x;
  int lane = tid & 31;
  int wave = tid >> 5;
  const __bf16* weff  = weffPk + (size_t)b * 32768;
  const float*  c1b   = c1 + (size_t)b * 256;

  for (int t0 = 0; t0 < TDIM; t0 += 16) {
    // ---- stage hist tile [16,128] via async global->LDS (16B per lane)
    {
      int r = tid >> 4, c = (tid & 15) * 8;
      int t = t0 + r; if (t > TDIM - 1) t = TDIM - 1;
      const __bf16* g = histBF + ((size_t)b * TDIM + t) * HC + c;
      uint32_t ldsOff = (uint32_t)(uintptr_t)(histT + r * HC + c);
      asm volatile("global_load_async_to_lds_b128 %0, %1, off"
                   :: "v"(ldsOff), "v"((unsigned long long)(uintptr_t)g)
                   : "memory");
      asm volatile("s_wait_asynccnt 0" ::: "memory");
    }
    __syncthreads();
    // ---- GEMM1 (folded): [16,128] @ W_eff[b][128,256] + c1[b], sigmoid
    for (int nt = wave; nt < 16; nt += 8) {
      v8f acc = {0.f, 0.f, 0.f, 0.f, 0.f, 0.f, 0.f, 0.f};
#pragma unroll
      for (int kc = 0; kc < 4; ++kc) {
        v16bf a  = load_a_frag(histT, HC, kc * 32, lane);
        v16bf bb = *(const v16bf*)(weff + (((size_t)kc * 16 + nt) * 32 + lane) * 16);
        acc = wmma_bf16(a, bb, acc);
      }
      int   n    = nt * 16 + (lane & 15);
      int   mb   = (lane >> 4) * 8;
      float bias = c1b[n];              // cur@(W1a+W1c) + b1, precomputed
#pragma unroll
      for (int r = 0; r < 8; ++r)
        h1T[(r + mb) * 256 + n] = (__bf16)sigmoidf_(acc[r] + bias);
    }
    __syncthreads();
    // ---- GEMM2: [16,256] @ [256,128], sigmoid -> h2 (bf16)
    {
      int nt = wave;
      v8f acc = {0.f, 0.f, 0.f, 0.f, 0.f, 0.f, 0.f, 0.f};
#pragma unroll
      for (int kc = 0; kc < 8; ++kc) {
        v16bf a  = load_a_frag(h1T, 256, kc * 32, lane);
        v16bf bb = *(const v16bf*)(w2pk + (((size_t)kc * 8 + nt) * 32 + lane) * 16);
        acc = wmma_bf16(a, bb, acc);
      }
      int   n    = nt * 16 + (lane & 15);
      int   mb   = (lane >> 4) * 8;
      float bias = b2[n];
#pragma unroll
      for (int r = 0; r < 8; ++r)
        h2T[(r + mb) * HC + n] = (__bf16)sigmoidf_(acc[r] + bias);
    }
    __syncthreads();
    // ---- GEMM3: [16,128] @ [128,1] -> attention logits
    if (tid < 16) {
      float s = b3[0];
      for (int k = 0; k < HC; ++k) s += (float)h2T[tid * HC + k] * w3[k];
      awL[t0 + tid] = s;
    }
    __syncthreads();
  }

  // ---- masked softmax over T (trivial; serialized on one lane)
  if (tid == 0) {
    float mx = -3.4e38f;
    for (int t = 0; t < TDIM; ++t) {
      float v = (maskG[(size_t)b * TDIM + t] == 1) ? awL[t] : NEGV;
      if (v > mx) mx = v;
    }
    float se = 0.f, ms = 0.f;
    for (int t = 0; t < TDIM; ++t) {
      int   m = maskG[(size_t)b * TDIM + t];
      float v = (m == 1) ? awL[t] : NEGV;
      float e = __expf(v - mx);
      se += e;
      ms += (float)m;
      awL[t] = e * (float)m;             // fold maskf into the numerator
    }
    float inv = 1.f / se;
    for (int t = 0; t < TDIM; ++t) awL[t] *= inv;
    sMsum = ms;
  }
  __syncthreads();

  // ---- attention pooling + masked mean + assemble x row [448]
  if (tid < HC) {
    float s = 0.f, hs = 0.f;
    for (int t = 0; t < TDIM; ++t) {
      float hv = (float)histBF[((size_t)b * TDIM + t) * HC + tid];
      s  += awL[t] * hv;
      hs += (float)maskG[(size_t)b * TDIM + t] * hv;
    }
    float* xr = x + (size_t)b * DMLP;
    xr[192 + tid] = s;
    xr[320 + tid] = hs / sMsum;
    xr[64 + tid]  = curFG[b * HC + tid];
    if (tid < 64) xr[tid] = userT[(size_t)uid[b] * EDIM + tid];
  }
}

// ---------------------------------------------------------------------------
// K3: per-column batch stats (mean, rsqrt(var+eps)) over M rows
// ---------------------------------------------------------------------------
__global__ void k_colstats(const float* __restrict__ src, int M, int C,
                           float* __restrict__ mean, float* __restrict__ inv) {
  __shared__ float s1[256], s2[256];
  int c = blockIdx.x, tid = threadIdx.x;
  float a = 0.f, q = 0.f;
  for (int r = tid; r < M; r += 256) {
    float v = src[(size_t)r * C + c];
    a += v; q += v * v;
  }
  s1[tid] = a; s2[tid] = q;
  __syncthreads();
  for (int s = 128; s > 0; s >>= 1) {
    if (tid < s) { s1[tid] += s1[tid + s]; s2[tid] += s2[tid + s]; }
    __syncthreads();
  }
  if (tid == 0) {
    float mu  = s1[0] / (float)M;
    float var = s2[0] / (float)M - mu * mu;
    mean[c] = mu;
    inv[c]  = rsqrtf(var + 1e-5f);
  }
}

// K4: BN apply (+ optional LeakyReLU via slope; slope=1 -> identity) -> bf16
__global__ void k_bn_apply(const float* __restrict__ src,
                           const float* __restrict__ mean,
                           const float* __restrict__ inv,
                           const float* __restrict__ g,
                           const float* __restrict__ beta, float slope,
                           int M, int C, __bf16* __restrict__ dst) {
  int i = blockIdx.x * blockDim.x + threadIdx.x;
  if (i >= M * C) return;
  int c = i % C;
  float v = (src[i] - mean[c]) * inv[c] * g[c] + beta[c];
  v = (v > 0.f) ? v : v * slope;
  dst[i] = (__bf16)v;
}

// ---------------------------------------------------------------------------
// K5: generic bf16 WMMA GEMM: C[M,N] = A[M,K] @ Wpk + bias, f32 out.
// Workgroup = (16-row M tile) x (128-col N slab), 8 waves = 8 n-tiles.
// ---------------------------------------------------------------------------
__global__ void __launch_bounds__(256) k_gemm(const __bf16* __restrict__ A,
                                              const __bf16* __restrict__ Wpk,
                                              const float* __restrict__ bias,
                                              float* __restrict__ C,
                                              int M, int K, int N) {
  __shared__ __bf16 at[16 * 512];
  int tid = threadIdx.x, lane = tid & 31, wave = tid >> 5;
  int m0 = blockIdx.x * 16;
  int nb = blockIdx.y * 8 + wave;
  for (int e = tid * 8; e < 16 * K; e += 2048) {
    int r = e / K, c = e % K;
    *(v8bf*)(at + e) = *(const v8bf*)(A + (size_t)(m0 + r) * K + c);
  }
  __syncthreads();
  int nT = N >> 4;
  v8f acc = {0.f, 0.f, 0.f, 0.f, 0.f, 0.f, 0.f, 0.f};
  for (int kc = 0; kc < (K >> 5); ++kc) {
    v16bf a  = load_a_frag(at, K, kc * 32, lane);
    v16bf bb = *(const v16bf*)(Wpk + (((size_t)kc * nT + nb) * 32 + lane) * 16);
    acc = wmma_bf16(a, bb, acc);
  }
  int   n  = nb * 16 + (lane & 15);
  int   mb = (lane >> 4) * 8;
  float bs = bias[n];
#pragma unroll
  for (int r = 0; r < 8; ++r)
    C[(size_t)(m0 + r + mb) * N + n] = acc[r] + bs;
}

// K6: final tiny layer [1024,256] @ [256,2] + b3 -> out
__global__ void k_final(const __bf16* __restrict__ a2,
                        const float* __restrict__ w3,
                        const float* __restrict__ b3,
                        float* __restrict__ out) {
  int i = blockIdx.x * blockDim.x + threadIdx.x;
  if (i >= BDIM * 2) return;
  int b = i >> 1, o = i & 1;
  float s = b3[o];
  for (int k = 0; k < 256; ++k)
    s += (float)a2[(size_t)b * 256 + k] * w3[k * 2 + o];
  out[i] = s;
}

// ---------------------------------------------------------------------------
extern "C" void kernel_launch(void* const* d_in, const int* in_sizes, int n_in,
                              void* d_out, int out_size, void* d_ws,
                              size_t ws_size, hipStream_t stream) {
  (void)in_sizes; (void)n_in; (void)out_size; (void)ws_size;
  const int*   uid    = (const int*)d_in[0];
  const int*   mid    = (const int*)d_in[1];
  const int*   cat    = (const int*)d_in[2];
  const int*   hmid   = (const int*)d_in[3];
  const int*   hcat   = (const int*)d_in[4];
  const int*   mask   = (const int*)d_in[5];
  const float* userT  = (const float*)d_in[6];
  const float* matT   = (const float*)d_in[7];
  const float* catT   = (const float*)d_in[8];
  const float* lau_w1 = (const float*)d_in[9];
  const float* lau_b1 = (const float*)d_in[10];
  const float* lau_w2 = (const float*)d_in[11];
  const float* lau_b2 = (const float*)d_in[12];
  const float* lau_w3 = (const float*)d_in[13];
  const float* lau_b3 = (const float*)d_in[14];
  const float* bn0_g  = (const float*)d_in[15];
  const float* bn0_b  = (const float*)d_in[16];
  const float* mlp_w1 = (const float*)d_in[17];
  const float* mlp_b1 = (const float*)d_in[18];
  const float* bn1_g  = (const float*)d_in[19];
  const float* bn1_b  = (const float*)d_in[20];
  const float* mlp_w2 = (const float*)d_in[21];
  const float* mlp_b2 = (const float*)d_in[22];
  const float* bn2_g  = (const float*)d_in[23];
  const float* bn2_b  = (const float*)d_in[24];
  const float* mlp_w3 = (const float*)d_in[25];
  const float* mlp_b3 = (const float*)d_in[26];
  float* out = (float*)d_out;

  char*  ws  = (char*)d_ws;
  size_t off = 0;
  auto alloc = [&](size_t bytes) -> void* {
    void* p = ws + off;
    off += (bytes + 255) & ~(size_t)255;
    return p;
  };
  __bf16* histBF = (__bf16*)alloc((size_t)BDIM * TDIM * HC * 2);   // 52 MB
  __bf16* weffPk = (__bf16*)alloc((size_t)BDIM * HC * 256 * 2);    // 67 MB
  __bf16* w1acPk = (__bf16*)alloc((size_t)HC * 256 * 2);
  __bf16* w2L    = (__bf16*)alloc((size_t)256 * 128 * 2);
  __bf16* w1M    = (__bf16*)alloc((size_t)DMLP * 512 * 2);
  __bf16* w2M    = (__bf16*)alloc((size_t)512 * 256 * 2);
  float*  curF   = (float*) alloc((size_t)BDIM * HC * 4);
  __bf16* curB   = (__bf16*)alloc((size_t)BDIM * HC * 2);
  float*  c1     = (float*) alloc((size_t)BDIM * 256 * 4);
  float*  x      = (float*) alloc((size_t)BDIM * DMLP * 4);
  __bf16* x0bf   = (__bf16*)alloc((size_t)BDIM * DMLP * 2);
  float*  y1     = (float*) alloc((size_t)BDIM * 512 * 4);
  __bf16* a1bf   = (__bf16*)alloc((size_t)BDIM * 512 * 2);
  float*  y2     = (float*) alloc((size_t)BDIM * 256 * 4);
  __bf16* a2bf   = (__bf16*)alloc((size_t)BDIM * 256 * 2);
  float*  mean   = (float*) alloc(512 * 4);
  float*  inv    = (float*) alloc(512 * 4);

  // 0) gathers -> bf16
  k_gather_hist<<<(BDIM * TDIM * 32 + 255) / 256, 256, 0, stream>>>(
      hmid, hcat, matT, catT, histBF);
  k_gather_cur<<<(BDIM * HC + 255) / 256, 256, 0, stream>>>(
      mid, cat, matT, catT, curF, curB);
  // 1) pack static weights; build folded per-batch LAU weights + bias
  k_pack_w1ac<<<(HC * 256 + 255) / 256, 256, 0, stream>>>(lau_w1, w1acPk);
  k_pack_w<<<(256 * 128 + 255) / 256, 256, 0, stream>>>(lau_w2, w2L, 256, 128);
  k_pack_w<<<(448 * 512 + 255) / 256, 256, 0, stream>>>(mlp_w1, w1M, 448, 512);
  k_pack_w<<<(512 * 256 + 255) / 256, 256, 0, stream>>>(mlp_w2, w2M, 512, 256);
  // c1[b,n] = cur[b] @ (W1a+W1c) + b1   (WMMA GEMM, M=1024 K=128 N=256)
  k_gemm<<<dim3(BDIM / 16, 256 / 128), 256, 0, stream>>>(
      curB, w1acPk, lau_b1, c1, BDIM, HC, 256);
  k_build_weff<<<(BDIM * HC * 256) / 256, 256, 0, stream>>>(lau_w1, curF, weffPk);
  // 2) fused LAU + softmax + pooling -> x [B,448]
  k_lau<<<BDIM, 256, 0, stream>>>(uid, mask, userT, histBF, curF, weffPk, c1,
                                  w2L, lau_b2, lau_w3, lau_b3, x);
  // 3) BN0 -> bf16
  k_colstats<<<DMLP, 256, 0, stream>>>(x, BDIM, DMLP, mean, inv);
  k_bn_apply<<<(BDIM * DMLP + 255) / 256, 256, 0, stream>>>(
      x, mean, inv, bn0_g, bn0_b, 1.0f, BDIM, DMLP, x0bf);
  // 4) MLP layer 1: [1024,448] @ [448,512]
  k_gemm<<<dim3(BDIM / 16, 512 / 128), 256, 0, stream>>>(
      x0bf, w1M, mlp_b1, y1, BDIM, DMLP, 512);
  k_colstats<<<512, 256, 0, stream>>>(y1, BDIM, 512, mean, inv);
  k_bn_apply<<<(BDIM * 512 + 255) / 256, 256, 0, stream>>>(
      y1, mean, inv, bn1_g, bn1_b, 0.1f, BDIM, 512, a1bf);
  // 5) MLP layer 2: [1024,512] @ [512,256]
  k_gemm<<<dim3(BDIM / 16, 256 / 128), 256, 0, stream>>>(
      a1bf, w2M, mlp_b2, y2, BDIM, 512, 256);
  k_colstats<<<256, 256, 0, stream>>>(y2, BDIM, 256, mean, inv);
  k_bn_apply<<<(BDIM * 256 + 255) / 256, 256, 0, stream>>>(
      y2, mean, inv, bn2_g, bn2_b, 0.1f, BDIM, 256, a2bf);
  // 6) final [1024,256] @ [256,2]
  k_final<<<(BDIM * 2 + 255) / 256, 256, 0, stream>>>(a2bf, mlp_w3, mlp_b3, out);
}
